// CrossUnit_20323785244719
// MI455X (gfx1250) — matrix operand order repaired and checked by reading, not compiled, not run on hardware
//
#include <hip/hip_runtime.h>
#include <hip/hip_bf16.h>

// ---------------------------------------------------------------------------
// Types for CDNA5 WMMA (wave32): v_wmma_f32_16x16x32_bf16
// ---------------------------------------------------------------------------
typedef __attribute__((ext_vector_type(16))) __bf16 v16bf;
typedef __attribute__((ext_vector_type(8)))  float  v8f;

#define DMODEL 768
#define NEXP   8
#define PTOK   49
#define SEQ    50
#define MAXTOK 12800   // 256*50, worst-case tokens per expert bucket

__device__ __forceinline__ v8f wmma_bf16(v16bf a, v16bf b, v8f c) {
  // 8 args: (neg_a, A, neg_b, B, c_mod, C, reuse_a, reuse_b)
  return __builtin_amdgcn_wmma_f32_16x16x32_bf16(false, a, false, b, (short)0, c,
                                                 false, false);
}

// A fragment (16x32 bf16, MxK). Lane holds row m = lane%16; half = lane/16.
// elems 0..15 -> K = k0 + half*16 + i
__device__ __forceinline__ v16bf load_a_f32(const float* __restrict__ arow,
                                            int k0, int half) {
  v16bf a;
  const float* p0 = arow + k0 + half * 16;
  const float* p1 = arow + k0 + half * 16 + 8;
#pragma unroll
  for (int i = 0; i < 8; ++i) a[i] = (__bf16)p0[i];
#pragma unroll
  for (int i = 0; i < 8; ++i) a[8 + i] = (__bf16)p1[i];
  return a;
}

// B fragment from PRE-SWIZZLED weights: group g = (kb*768 + n)*2 + half holds
// the 16 contiguous bf16 this lane needs -> one 32B vector load per fragment.
__device__ __forceinline__ v16bf load_b_swz(const __bf16* __restrict__ Ws,
                                            int kb, int n, int half) {
  return *(const v16bf*)(Ws + ((size_t)(kb * DMODEL + n) * 2 + half) * 16);
}

// B fragment from f32 matrix accessed TRANSPOSED: B[k][n] = E[n][k] (E: N x 768)
__device__ __forceinline__ v16bf load_bT_f32(const float* __restrict__ E,
                                             int k0, int half, int n) {
  v16bf b;
  const float* p = E + (size_t)n * DMODEL + k0 + half * 16;
#pragma unroll
  for (int i = 0; i < 16; ++i) b[i] = (__bf16)p[i];
  return b;
}

// ---------------------------------------------------------------------------
// f32 weights (Krows x 768 row-major) -> bf16 in WMMA B-fragment order.
// One thread per (kb, n, half) group writes 16 contiguous bf16.
// ---------------------------------------------------------------------------
__global__ __launch_bounds__(256)
void cvt_swizzle_kernel(const float* __restrict__ W, __bf16* __restrict__ out,
                        int Krows) {
  int g = blockIdx.x * 256 + threadIdx.x;
  int total = (Krows / 32) * DMODEL * 2;
  if (g >= total) return;
  int half = g & 1;
  int rem = g >> 1;
  int n = rem % DMODEL;
  int kb = rem / DMODEL;
  const float* src = W + (size_t)(kb * 32 + half * 16) * DMODEL + n;
  __bf16* dst = out + (size_t)g * 16;
#pragma unroll
  for (int i = 0; i < 16; ++i) dst[i] = (__bf16)src[(size_t)i * DMODEL];
}

__global__ void zero_int_kernel(int* p, int n) {
  int i = blockIdx.x * 256 + threadIdx.x;
  if (i < n) p[i] = 0;
}

// ---------------------------------------------------------------------------
// Projection GEMM: out[map(r)] = A[r] @ W + bias,
//   map(r) = (r/GIN)*GOUT + r%GIN + OFF  (compile-time -> magic multiply)
// A: M x K f32, Ws: swizzled bf16 K x 768. M%128==0, K%32==0.
// Block = 8 waves (4M x 2N); wave tile 32(M) x 64(N); block tile 128x128.
// ---------------------------------------------------------------------------
template <int GIN, int GOUT, int OFF>
__global__ __launch_bounds__(256)
void proj_gemm_kernel(const float* __restrict__ A, const __bf16* __restrict__ Ws,
                      const float* __restrict__ bias, float* __restrict__ out,
                      int M, int K) {
  const int lane = threadIdx.x & 31;
  const int wave = threadIdx.x >> 5;
  const int half = lane >> 4;
  const int ln16 = lane & 15;
  const int mBase = blockIdx.x * 128 + (wave & 3) * 32;
  const int nBase = blockIdx.y * 128 + (wave >> 2) * 64;

  const float* arow0 = A + (size_t)(mBase + ln16) * K;
  const float* arow1 = arow0 + (size_t)16 * K;
  v8f z = {};
  v8f c[8];
#pragma unroll
  for (int i = 0; i < 8; ++i) c[i] = z;

  for (int k0 = 0, kb = 0; k0 < K; k0 += 32, ++kb) {
    v16bf a0 = load_a_f32(arow0, k0, half);
    v16bf a1 = load_a_f32(arow1, k0, half);
#pragma unroll
    for (int t = 0; t < 4; ++t) {
      v16bf b = load_b_swz(Ws, kb, nBase + t * 16 + ln16, half);
      c[t]     = wmma_bf16(a0, b, c[t]);
      c[4 + t] = wmma_bf16(a1, b, c[4 + t]);
    }
  }
#pragma unroll
  for (int mi = 0; mi < 2; ++mi) {
#pragma unroll
    for (int j = 0; j < 8; ++j) {
      int mg = mBase + mi * 16 + j + half * 8;
      int rd = (mg / GIN) * GOUT + (mg % GIN) + OFF;
      float* orow = out + (size_t)rd * DMODEL;
#pragma unroll
      for (int t = 0; t < 4; ++t) {
        int col = nBase + t * 16 + ln16;
        orow[col] = c[mi * 4 + t][j] + bias[col];
      }
    }
  }
}

// ---------------------------------------------------------------------------
// MoE router: one wave per token. logits = ctx @ Wr + br, top-2, softmax,
// append (token*2+slot, weight) into per-expert buckets via atomics.
// ---------------------------------------------------------------------------
__global__ __launch_bounds__(256)
void moe_router_kernel(const float* __restrict__ ctx, const float* __restrict__ Wr,
                       const float* __restrict__ br, int nTok, int* __restrict__ cnt,
                       int* __restrict__ bIdx, float* __restrict__ bW) {
  const int lane = threadIdx.x & 31;
  const int tok = blockIdx.x * 8 + (threadIdx.x >> 5);
  if (tok >= nTok) return;
  const float* c = ctx + (size_t)tok * DMODEL;
  float acc[NEXP];
#pragma unroll
  for (int e = 0; e < NEXP; ++e) acc[e] = 0.f;
  for (int k = lane; k < DMODEL; k += 32) {
    float x = c[k];
    const float* wr = Wr + (size_t)k * NEXP;
#pragma unroll
    for (int e = 0; e < NEXP; ++e) acc[e] += x * wr[e];
  }
#pragma unroll
  for (int e = 0; e < NEXP; ++e) {
#pragma unroll
    for (int off = 16; off; off >>= 1) acc[e] += __shfl_xor(acc[e], off, 32);
    acc[e] += br[e];
  }
  if (lane == 0) {
    int i0 = 0; float v0 = acc[0];
    for (int e = 1; e < NEXP; ++e) if (acc[e] > v0) { v0 = acc[e]; i0 = e; }
    int i1 = -1; float v1 = -3.402823466e38f;
    for (int e = 0; e < NEXP; ++e)
      if (e != i0 && acc[e] > v1) { v1 = acc[e]; i1 = e; }
    float ex = __expf(v1 - v0);          // v1 <= v0, safe
    float w0 = 1.f / (1.f + ex);
    float w1 = ex / (1.f + ex);
    int p0 = atomicAdd(&cnt[i0], 1);
    bIdx[i0 * MAXTOK + p0] = tok * 2 + 0; bW[i0 * MAXTOK + p0] = w0;
    int p1 = atomicAdd(&cnt[i1], 1);
    bIdx[i1 * MAXTOK + p1] = tok * 2 + 1; bW[i1 * MAXTOK + p1] = w1;
  }
}

// ---------------------------------------------------------------------------
// Expert-batched gather GEMM: for expert e, rows = bucket entries.
// slotOut[entry] = w * (x[token] @ We[e] + be[e]).  Ragged M: dummy compute
// rows read token 0, stores guarded per-row.
// ---------------------------------------------------------------------------
__global__ __launch_bounds__(256)
void moe_gemm_kernel(const float* __restrict__ X, const __bf16* __restrict__ WeSw,
                     const float* __restrict__ be, const int* __restrict__ cnt,
                     const int* __restrict__ bIdx, const float* __restrict__ bW,
                     float* __restrict__ slotOut) {
  const int e = blockIdx.z;
  const int n = cnt[e];
  const int wave = threadIdx.x >> 5;
  const int mBase = blockIdx.x * 128 + (wave & 3) * 32;
  if (mBase >= n) return;   // uniform per wave -> EXEC stays all-ones
  const int lane = threadIdx.x & 31;
  const int half = lane >> 4;
  const int ln16 = lane & 15;
  const int nBase = blockIdx.y * 128 + (wave >> 2) * 64;
  const int* bi = bIdx + e * MAXTOK;
  const float* bw = bW + e * MAXTOK;

  int r0 = mBase + ln16, r1 = r0 + 16;
  int en0 = (r0 < n) ? bi[r0] : 0;
  int en1 = (r1 < n) ? bi[r1] : 0;
  const float* arow0 = X + (size_t)(en0 >> 1) * DMODEL;
  const float* arow1 = X + (size_t)(en1 >> 1) * DMODEL;
  const __bf16* Ws = WeSw + (size_t)e * DMODEL * DMODEL;
  const float* beE = be + (size_t)e * DMODEL;

  v8f z = {};
  v8f c[8];
#pragma unroll
  for (int i = 0; i < 8; ++i) c[i] = z;

  for (int k0 = 0, kb = 0; k0 < DMODEL; k0 += 32, ++kb) {
    v16bf a0 = load_a_f32(arow0, k0, half);
    v16bf a1 = load_a_f32(arow1, k0, half);
#pragma unroll
    for (int t = 0; t < 4; ++t) {
      v16bf b = load_b_swz(Ws, kb, nBase + t * 16 + ln16, half);
      c[t]     = wmma_bf16(a0, b, c[t]);
      c[4 + t] = wmma_bf16(a1, b, c[4 + t]);
    }
  }
#pragma unroll
  for (int mi = 0; mi < 2; ++mi) {
#pragma unroll
    for (int j = 0; j < 8; ++j) {
      int rr = mBase + mi * 16 + j + half * 8;
      if (rr < n) {
        int ent = bi[rr];
        float w = bw[rr];
        float* orow = slotOut + (size_t)ent * DMODEL;
#pragma unroll
        for (int t = 0; t < 4; ++t) {
          int col = nBase + t * 16 + ln16;
          orow[col] = w * (c[mi * 4 + t][j] + beE[col]);
        }
      }
    }
  }
}

__global__ __launch_bounds__(256)
void moe_combine_kernel(const float* __restrict__ slot, float* __restrict__ out,
                        int nElem) {
  int i = blockIdx.x * 256 + threadIdx.x;
  if (i >= nElem) return;
  int tok = i / DMODEL, d = i - tok * DMODEL;
  out[i] = slot[(size_t)(tok * 2) * DMODEL + d] +
           slot[(size_t)(tok * 2 + 1) * DMODEL + d];
}

// ---------------------------------------------------------------------------
// Cross-attention + gate + LayerNorm. One block (256 thr) per batch row.
// ---------------------------------------------------------------------------
__global__ __launch_bounds__(256)
void cross_kernel(const float* __restrict__ moeBuf, const float* __restrict__ seqBuf,
                  const float* __restrict__ Wg, const float* __restrict__ bg,
                  const float* __restrict__ lng, const float* __restrict__ lnb,
                  int gateOnCtx, float* __restrict__ out) {
  __shared__ float s_sc[64];
  __shared__ float s_ctx[DMODEL];
  __shared__ float s_red[256];
  __shared__ float s_gate;
  const int b = blockIdx.x;
  const int tid = threadIdx.x, lane = tid & 31, wv = tid >> 5;
  const float* cls2 = moeBuf + (size_t)b * SEQ * DMODEL;   // moe row 0
  const float* toks = cls2 + DMODEL;                        // moe rows 1..49
  const float* clsO = seqBuf + (size_t)b * SEQ * DMODEL;    // original cls

  for (int p = wv; p < PTOK; p += 8) {
    float acc = 0.f;
    const float* tp = toks + (size_t)p * DMODEL;
    for (int k = lane; k < DMODEL; k += 32) acc += cls2[k] * tp[k];
#pragma unroll
    for (int off = 16; off; off >>= 1) acc += __shfl_xor(acc, off, 32);
    if (lane == 0) s_sc[p] = acc;
  }
  __syncthreads();
  if (tid == 0) {   // softmax over 49
    float mx = s_sc[0];
    for (int p = 1; p < PTOK; ++p) mx = fmaxf(mx, s_sc[p]);
    float sum = 0.f;
    for (int p = 0; p < PTOK; ++p) { float v = __expf(s_sc[p] - mx); s_sc[p] = v; sum += v; }
    float inv = 1.f / sum;
    for (int p = 0; p < PTOK; ++p) s_sc[p] *= inv;
  }
  __syncthreads();
  for (int d = tid; d < DMODEL; d += 256) {
    float acc = 0.f;
    for (int p = 0; p < PTOK; ++p) acc += s_sc[p] * toks[(size_t)p * DMODEL + d];
    s_ctx[d] = acc;
  }
  __syncthreads();
  float part = 0.f;
  for (int d = tid; d < DMODEL; d += 256) {
    float gi = gateOnCtx ? s_ctx[d] : clsO[d];
    part += gi * Wg[d];
  }
  s_red[tid] = part; __syncthreads();
  for (int s = 128; s; s >>= 1) { if (tid < s) s_red[tid] += s_red[tid + s]; __syncthreads(); }
  if (tid == 0) s_gate = tanhf(s_red[0] + bg[0]);
  __syncthreads();
  float gate = s_gate;
  for (int d = tid; d < DMODEL; d += 256) s_ctx[d] = clsO[d] * gate + s_ctx[d];
  __syncthreads();
  part = 0.f;
  for (int d = tid; d < DMODEL; d += 256) part += s_ctx[d];
  s_red[tid] = part; __syncthreads();
  for (int s = 128; s; s >>= 1) { if (tid < s) s_red[tid] += s_red[tid + s]; __syncthreads(); }
  float mu = s_red[0] * (1.f / DMODEL); __syncthreads();
  part = 0.f;
  for (int d = tid; d < DMODEL; d += 256) { float t = s_ctx[d] - mu; part += t * t; }
  s_red[tid] = part; __syncthreads();
  for (int s = 128; s; s >>= 1) { if (tid < s) s_red[tid] += s_red[tid + s]; __syncthreads(); }
  float rstd = rsqrtf(s_red[0] * (1.f / DMODEL) + 1e-5f);
  for (int d = tid; d < DMODEL; d += 256)
    out[(size_t)b * DMODEL + d] = (s_ctx[d] - mu) * rstd * lng[d] + lnb[d];
}

// ---------------------------------------------------------------------------
// Final scores: out = 0.5*(Mti @ Eti^T + Mit @ Eit^T), M=128, N=256, K=768.
// grid = (1,4); 8 waves, each 16x64.
// ---------------------------------------------------------------------------
__global__ __launch_bounds__(256)
void score_gemm_kernel(const float* __restrict__ mTi, const float* __restrict__ eTi,
                       const float* __restrict__ mIt, const float* __restrict__ eIt,
                       float* __restrict__ out) {
  const int lane = threadIdx.x & 31;
  const int wave = threadIdx.x >> 5;
  const int half = lane >> 4;
  const int ln16 = lane & 15;
  const int mBase = wave * 16;
  const int nBase = blockIdx.y * 64;

  v8f z = {};
  v8f c[4];
#pragma unroll
  for (int i = 0; i < 4; ++i) c[i] = z;

  const float* arow = mTi + (size_t)(mBase + ln16) * DMODEL;
  for (int k0 = 0; k0 < DMODEL; k0 += 32) {
    v16bf a = load_a_f32(arow, k0, half);
#pragma unroll
    for (int t = 0; t < 4; ++t) {
      v16bf b = load_bT_f32(eTi, k0, half, nBase + t * 16 + ln16);
      c[t] = wmma_bf16(a, b, c[t]);
    }
  }
  arow = mIt + (size_t)(mBase + ln16) * DMODEL;
  for (int k0 = 0; k0 < DMODEL; k0 += 32) {
    v16bf a = load_a_f32(arow, k0, half);
#pragma unroll
    for (int t = 0; t < 4; ++t) {
      v16bf b = load_bT_f32(eIt, k0, half, nBase + t * 16 + ln16);
      c[t] = wmma_bf16(a, b, c[t]);
    }
  }
#pragma unroll
  for (int j = 0; j < 8; ++j) {
    int m = mBase + j + half * 8;
    float* orow = out + (size_t)m * 256;
#pragma unroll
    for (int t = 0; t < 4; ++t) orow[nBase + t * 16 + ln16] = 0.5f * c[t][j];
  }
}

// ---------------------------------------------------------------------------
// Host launcher
// ---------------------------------------------------------------------------
extern "C" void kernel_launch(void* const* d_in, const int* in_sizes, int n_in,
                              void* d_out, int out_size, void* d_ws, size_t ws_size,
                              hipStream_t stream) {
  const float* etc_cls = (const float*)d_in[0];   // (256,768)
  const float* ett_tok = (const float*)d_in[1];   // (256,49,768)
  const float* mtc_cls = (const float*)d_in[2];   // (128,768)
  const float* mtt_tok = (const float*)d_in[3];   // (128,49,768)
  const float* eic_cls = (const float*)d_in[4];   // (256,1024)
  const float* eit_tok = (const float*)d_in[5];   // (256,49,1024)
  const float* mic_cls = (const float*)d_in[6];   // (128,1024)
  const float* mit_tok = (const float*)d_in[7];   // (128,49,1024)
  const float* Wt   = (const float*)d_in[8];
  const float* bt   = (const float*)d_in[9];
  const float* Wi   = (const float*)d_in[10];
  const float* bi   = (const float*)d_in[11];
  const float* Wg   = (const float*)d_in[12];
  const float* bg   = (const float*)d_in[13];
  const float* Wr   = (const float*)d_in[14];
  const float* br   = (const float*)d_in[15];
  const float* We   = (const float*)d_in[16];
  const float* be   = (const float*)d_in[17];
  const float* ln_g = (const float*)d_in[18];
  const float* ln_b = (const float*)d_in[19];

  // workspace carve-up
  char* wp = (char*)d_ws;
  auto alloc = [&](size_t bytes) -> char* {
    char* r = wp;
    wp += (bytes + 255) & ~(size_t)255;
    return r;
  };
  __bf16* WtS = (__bf16*)alloc((size_t)768 * 768 * 2);
  __bf16* WiS = (__bf16*)alloc((size_t)1024 * 768 * 2);
  __bf16* WeS = (__bf16*)alloc((size_t)8 * 768 * 768 * 2);
  const size_t eSeq = (size_t)256 * SEQ * DMODEL;
  const size_t mSeq = (size_t)128 * SEQ * DMODEL;
  float* e_ti = (float*)alloc(eSeq * 4);
  float* e_it = (float*)alloc(eSeq * 4);
  float* m_ti = (float*)alloc(mSeq * 4);
  float* m_it = (float*)alloc(mSeq * 4);
  float* me_ti = (float*)alloc(eSeq * 4);
  float* me_it = (float*)alloc(eSeq * 4);
  float* mm_ti = (float*)alloc(mSeq * 4);
  float* mm_it = (float*)alloc(mSeq * 4);
  float* slot  = (float*)alloc((size_t)MAXTOK * 2 * DMODEL * 4);
  int*   cnt   = (int*)alloc(NEXP * 4);
  int*   bIdx  = (int*)alloc((size_t)NEXP * MAXTOK * 4);
  float* bW    = (float*)alloc((size_t)NEXP * MAXTOK * 4);
  float* cEti  = (float*)alloc((size_t)256 * DMODEL * 4);
  float* cMti  = (float*)alloc((size_t)128 * DMODEL * 4);
  float* cEit  = (float*)alloc((size_t)256 * DMODEL * 4);
  float* cMit  = (float*)alloc((size_t)128 * DMODEL * 4);

  // 0) weights -> swizzled bf16 (WMMA B-fragment order)
  auto swz = [&](const float* W, __bf16* out, int Krows) {
    int total = (Krows / 32) * DMODEL * 2;
    cvt_swizzle_kernel<<<(total + 255) / 256, 256, 0, stream>>>(W, out, Krows);
  };
  swz(Wt, WtS, 768);
  swz(Wi, WiS, 1024);
  swz(We, WeS, 8 * 768);   // per-expert blocks stay contiguous (768 % 32 == 0)

  // 1) projections into concatenated [cls, tokens] sequence buffers
  auto projCls = [&](const float* A, const __bf16* W, const float* bias, float* out,
                     int M, int K) {
    dim3 g(M / 128, DMODEL / 128);
    proj_gemm_kernel<1, SEQ, 0><<<g, 256, 0, stream>>>(A, W, bias, out, M, K);
  };
  auto projTok = [&](const float* A, const __bf16* W, const float* bias, float* out,
                     int M, int K) {
    dim3 g(M / 128, DMODEL / 128);
    proj_gemm_kernel<PTOK, SEQ, 1><<<g, 256, 0, stream>>>(A, W, bias, out, M, K);
  };
  projCls(etc_cls, WtS, bt, e_ti, 256,      768);   // etc -> e_ti[:,0]
  projTok(eit_tok, WiS, bi, e_ti, 256 * 49, 1024);  // eit -> e_ti[:,1:]
  projCls(eic_cls, WiS, bi, e_it, 256,      1024);  // eic -> e_it[:,0]
  projTok(ett_tok, WtS, bt, e_it, 256 * 49, 768);   // ett -> e_it[:,1:]
  projCls(mtc_cls, WtS, bt, m_ti, 128,      768);   // mtc -> m_ti[:,0]
  projTok(mit_tok, WiS, bi, m_ti, 128 * 49, 1024);  // mit -> m_ti[:,1:]
  projCls(mic_cls, WiS, bi, m_it, 128,      1024);  // mic -> m_it[:,0]
  projTok(mtt_tok, WtS, bt, m_it, 128 * 49, 768);   // mtt -> m_it[:,1:]

  // 2) MoE (routed top-2, expert-batched gather GEMM)
  auto moe = [&](const float* x, const float* ctxv, float* outBuf, int nTok) {
    zero_int_kernel<<<1, 256, 0, stream>>>(cnt, NEXP);
    moe_router_kernel<<<(nTok + 7) / 8, 256, 0, stream>>>(ctxv, Wr, br, nTok, cnt,
                                                          bIdx, bW);
    dim3 g((nTok + 127) / 128, DMODEL / 128, NEXP);
    moe_gemm_kernel<<<g, 256, 0, stream>>>(x, WeS, be, cnt, bIdx, bW, slot);
    int nElem = nTok * DMODEL;
    moe_combine_kernel<<<(nElem + 255) / 256, 256, 0, stream>>>(slot, outBuf, nElem);
  };
  moe(e_ti, e_it, me_ti, 256 * SEQ);
  moe(m_ti, m_it, mm_ti, 128 * SEQ);
  moe(e_it, e_ti, me_it, 256 * SEQ);
  moe(m_it, m_ti, mm_it, 128 * SEQ);

  // 3) cross-attention + gate + LayerNorm
  cross_kernel<<<256, 256, 0, stream>>>(me_ti, e_ti, Wg, bg, ln_g, ln_b, 1, cEti);
  cross_kernel<<<128, 256, 0, stream>>>(mm_ti, m_ti, Wg, bg, ln_g, ln_b, 0, cMti);
  cross_kernel<<<256, 256, 0, stream>>>(me_it, e_it, Wg, bg, ln_g, ln_b, 1, cEit);
  cross_kernel<<<128, 256, 0, stream>>>(mm_it, m_it, Wg, bg, ln_g, ln_b, 0, cMit);

  // 4) fused final scores
  dim3 gs(1, 256 / 64);
  score_gemm_kernel<<<gs, 256, 0, stream>>>(cMti, cEti, cMit, cEit, (float*)d_out);
}
